// SlowFluidNet_46858093199469
// MI455X (gfx1250) — matrix operand order repaired and checked by reading, not compiled, not run on hardware
//
#include <hip/hip_runtime.h>
#include <stdint.h>

#define NN 2048
#define MM 2048
#define H0 18       // first hidden width
#define AST 24      // padded row stride (floats) for Af/As -> 96B rows, 16B-aligned slices
#define WAVES 8     // 256-thread block = 8 wave32
#define UNROLL 2    // independent chunks in flight per wave (fills WMMA hazard slots)

typedef __attribute__((ext_vector_type(16))) _Float16 v16h;
typedef __attribute__((ext_vector_type(8)))  _Float16 v8h;
typedef __attribute__((ext_vector_type(8)))  float    v8f;

static __device__ __forceinline__ float fast_tanh(float x) {
#if __has_builtin(__builtin_amdgcn_tanhf)
  return __builtin_amdgcn_tanhf(x);        // v_tanh_f32 on gfx1250
#elif __has_builtin(__builtin_amdgcn_tanh_f32)
  return __builtin_amdgcn_tanh_f32(x);
#else
  return tanhf(x);
#endif
}

// ---------------------------------------------------------------------------
// Kernel 1: per-m separable layer-0 partials (stride-24 padded rows).
// ---------------------------------------------------------------------------
__global__ void fluidnet_pre(const float* __restrict__ cur,
                             const float* __restrict__ fW0,
                             const float* __restrict__ sW0,
                             float* __restrict__ Af, float* __restrict__ As,
                             uint8_t* __restrict__ pt8) {
  int t = blockIdx.x * blockDim.x + threadIdx.x;
  if (t >= MM * H0) return;
  int m = t / H0, k = t - m * H0;
  const float* c = cur + m * 7;
  float af = 0.f, as = 0.f;
#pragma unroll
  for (int j = 0; j < 3; ++j) {
    af += c[j] * fW0[j * H0 + k] + c[3 + j] * fW0[(3 + j) * H0 + k];
    as += c[j] * sW0[j * H0 + k];
  }
  Af[m * AST + k] = af;
  As[m * AST + k] = as;
  if (k == 0) pt8[m] = (uint8_t)c[6];
}

// ---------------------------------------------------------------------------
// Kernel 2: one workgroup per center n; each wave owns pairs of 16-particle
// chunks (two independent WMMA streams interleave to cover the f16-WMMA
// RAW hazard window). 8 x v_wmma_f32_16x16x32_f16 per loop body.
// ---------------------------------------------------------------------------
__global__ __launch_bounds__(256) void fluidnet_main(
    const uint8_t* __restrict__ mask,     // [NN][MM] bool bytes
    const float*   __restrict__ center,   // [NN][6]
    const float* __restrict__ fW0, const float* __restrict__ fb0,
    const float* __restrict__ fW1, const float* __restrict__ fb1,
    const float* __restrict__ fW2, const float* __restrict__ fb2,
    const float* __restrict__ fW3, const float* __restrict__ fb3,
    const float* __restrict__ sW0, const float* __restrict__ sb0,
    const float* __restrict__ sW1, const float* __restrict__ sb1,
    const float* __restrict__ sW2, const float* __restrict__ sb2,
    const float* __restrict__ sW3, const float* __restrict__ sb3,
    const float* __restrict__ Af, const float* __restrict__ As,
    const uint8_t* __restrict__ pt8,
    float* __restrict__ out) {
  const int n    = blockIdx.x;
  const int tid  = threadIdx.x;
  const int wave = tid >> 5;
  const int lane = tid & 31;
  const int hh   = lane >> 4;   // lane half (A/B/C layout selector)
  const int col  = lane & 15;

  __shared__ float Bf_sh[H0], Bs_sh[H0];
  __shared__ float pred_sh[3];
  __shared__ __attribute__((aligned(16))) _Float16 tiles[UNROLL * WAVES][16][32];

  // per-n layer-0 partials: -pos_c@W0[t-rows] + vel_c@W0[vel-rows] + b0
  if (tid < H0) {
    const float* cc = center + (size_t)n * 6;
    float bf = fb0[tid], bs = sb0[tid];
#pragma unroll
    for (int j = 0; j < 3; ++j) {
      bf += cc[3 + j] * fW0[(6 + j) * H0 + tid] - cc[j] * fW0[j * H0 + tid];
      bs += cc[3 + j] * sW0[(3 + j) * H0 + tid] - cc[j] * sW0[j * H0 + tid];
    }
    Bf_sh[tid] = bf;
    Bs_sh[tid] = bs;
  }
  if (tid < 3) pred_sh[tid] = 0.f;
  {  // one-time zero so un-initialized LDS can never inject NaN into WMMA
    uint32_t* z = (uint32_t*)tiles;
    for (int i = tid; i < (int)(sizeof(tiles) / 4); i += 256) z[i] = 0u;
  }
  __syncthreads();

  _Float16(*__restrict__ tileC0)[32] = tiles[wave];           // chunk-0 h2/h3 tile
  _Float16(*__restrict__ tileC1)[32] = tiles[WAVES + wave];   // chunk-1 h2/h3 tile

  // B (weight) fragments, documented layout: lane holds N=col, K=((h<8)?h:h+8)+8*hh.
  v16h Bw1f, Bw1s, Bw2, Bw3;
#pragma unroll
  for (int h = 0; h < 16; ++h) {
    const int K = ((h < 8) ? h : h + 8) + 8 * hh;
    _Float16 w1f = (_Float16)0, w1s = (_Float16)0, w2 = (_Float16)0, w3 = (_Float16)0;
    if (K < H0 && col < 9) {
      w1f = (_Float16)fW1[K * 9 + col];
      w1s = (_Float16)sW1[K * 9 + col];
    }
    if (K < 9 && col < 6)                                w2 = (_Float16)fW2[K * 6 + col];
    else if (K >= 16 && K < 25 && col >= 8 && col < 14)  w2 = (_Float16)sW2[(K - 16) * 6 + (col - 8)];
    if (K < 6 && col < 3)                                w3 = (_Float16)fW3[K * 3 + col];
    else if (K >= 16 && K < 22 && col >= 8 && col < 11)  w3 = (_Float16)sW3[(K - 16) * 3 + (col - 8)];
    Bw1f[h] = w1f; Bw1s[h] = w1s; Bw2[h] = w2; Bw3[h] = w3;
  }

  // Chunk-invariant per-n partials in A-fragment element order.
  float bfK[16], bsK[16];
#pragma unroll
  for (int h = 0; h < 16; ++h) {
    const int K = ((h < 8) ? h : h + 8) + 8 * hh;
    const bool valid = (K < H0);
    bfK[h] = valid ? Bf_sh[K] : 0.f;
    bsK[h] = valid ? Bs_sh[K] : 0.f;
  }

  const float b1f = (col < 9) ? fb1[col] : 0.f;
  const float b1s = (col < 9) ? sb1[col] : 0.f;
  const float b2  = (col < 6) ? fb2[col] : ((col >= 8 && col < 14) ? sb2[col - 8] : 0.f);
  const float b3  = (col < 3) ? fb3[col] : ((col >= 8 && col < 11) ? sb3[col - 8] : 0.f);

  // Induction pointers (advance by constant strides; no per-iteration muls).
  const float*    afp = Af + (size_t)(wave * 16 * UNROLL + col) * AST;
  const float*    asp = As + (size_t)(wave * 16 * UNROLL + col) * AST;
  const uint64_t* mp  = (const uint64_t*)(mask + (size_t)n * MM) + wave * 2 * UNROLL;
  const uint64_t* pp  = (const uint64_t*)pt8 + wave * 2 * UNROLL;
  const size_t APITCH = (size_t)WAVES * 16 * UNROLL * AST;  // floats per iteration
  const int    MPITCH = WAVES * 2 * UNROLL;                 // u64 words per iteration

  const v8f zeroC = {};
  float acc = 0.f;
  int   cnt = 0;

  for (int it = 0; it < (MM / 16) / (WAVES * UNROLL); ++it) {
    __builtin_prefetch(afp + APITCH, 0, 1);   // global_prefetch_b8
    __builtin_prefetch(asp + APITCH, 0, 1);

    // ---- layer-1 A fragments for both chunks, built in registers ---------
    v16h A1f[UNROLL], A1s[UNROLL];
#pragma unroll
    for (int u = 0; u < UNROLL; ++u) {
      const float* afr = afp + (size_t)u * 16 * AST;
      const float* asr = asp + (size_t)u * 16 * AST;
      const float4 fa0 = *(const float4*)(afr + 8 * hh);
      const float4 fa1 = *(const float4*)(afr + 8 * hh + 4);
      const float4 sa0 = *(const float4*)(asr + 8 * hh);
      const float4 sa1 = *(const float4*)(asr + 8 * hh + 4);
      float2 fex = make_float2(0.f, 0.f), sex = make_float2(0.f, 0.f);
      if (hh == 0) {
        fex = *(const float2*)(afr + 16);
        sex = *(const float2*)(asr + 16);
      }
      const float fv[10] = {fa0.x, fa0.y, fa0.z, fa0.w, fa1.x, fa1.y, fa1.z, fa1.w, fex.x, fex.y};
      const float sv[10] = {sa0.x, sa0.y, sa0.z, sa0.w, sa1.x, sa1.y, sa1.z, sa1.w, sex.x, sex.y};
#pragma unroll
      for (int h = 0; h < 16; ++h) {
        const bool valid = (h < 8) || (hh == 0 && h < 10);
        A1f[u][h] = valid ? (_Float16)fast_tanh(fv[(h < 10) ? h : 0] + bfK[h]) : (_Float16)0;
        A1s[u][h] = valid ? (_Float16)fast_tanh(sv[(h < 10) ? h : 0] + bsK[h]) : (_Float16)0;
      }
    }

    v8f Df[UNROLL], Ds[UNROLL];
#pragma unroll
    for (int u = 0; u < UNROLL; ++u) {
      Df[u] = __builtin_amdgcn_wmma_f32_16x16x32_f16(false, A1f[u], false, Bw1f, (short)0, zeroC, false, false);
      Ds[u] = __builtin_amdgcn_wmma_f32_16x16x32_f16(false, A1s[u], false, Bw1s, (short)0, zeroC, false, false);
    }

    // ---- h2 = tanh(D + b1): fluid -> K 0..8, solid -> K 16..24 -----------
    if (col < 9) {
#pragma unroll
      for (int r = 0; r < 8; ++r) {
        const int p = r + 8 * hh;  // D layout: M = r + 8*hh, N = col
        tileC0[p][col]      = (_Float16)fast_tanh(Df[0][r] + b1f);
        tileC0[p][16 + col] = (_Float16)fast_tanh(Ds[0][r] + b1s);
        tileC1[p][col]      = (_Float16)fast_tanh(Df[1][r] + b1f);
        tileC1[p][16 + col] = (_Float16)fast_tanh(Ds[1][r] + b1s);
      }
    }

    v8h lo0 = *(const v8h*)&tileC0[col][8 * hh];
    v8h hi0 = *(const v8h*)&tileC0[col][16 + 8 * hh];
    v8h lo1 = *(const v8h*)&tileC1[col][8 * hh];
    v8h hi1 = *(const v8h*)&tileC1[col][16 + 8 * hh];
    v16h A2_0 = __builtin_shufflevector(lo0, hi0, 0,1,2,3,4,5,6,7,8,9,10,11,12,13,14,15);
    v16h A2_1 = __builtin_shufflevector(lo1, hi1, 0,1,2,3,4,5,6,7,8,9,10,11,12,13,14,15);
    v8f D2[UNROLL];
    D2[0] = __builtin_amdgcn_wmma_f32_16x16x32_f16(false, A2_0, false, Bw2, (short)0, zeroC, false, false);
    D2[1] = __builtin_amdgcn_wmma_f32_16x16x32_f16(false, A2_1, false, Bw2, (short)0, zeroC, false, false);

    // ---- h3 write with mask folded into the rows -------------------------
    uint64_t fm[UNROLL], sm[UNROLL];
#pragma unroll
    for (int u = 0; u < UNROLL; ++u) {
      const uint64_t mv = mp[u * 2 + hh];  // 8 mask bytes (values 0/1)
      const uint64_t pv = pp[u * 2 + hh];  // 8 ptype bytes (values 0/1)
      fm[u] = mv & ~pv;
      sm[u] = mv & pv;
    }

    if (col < 6) {
#pragma unroll
      for (int r = 0; r < 8; ++r) {
        const int p = r + 8 * hh;
        const float f0 = (float)((uint32_t)(fm[0] >> (8 * r)) & 1u);
        const float f1 = (float)((uint32_t)(fm[1] >> (8 * r)) & 1u);
        tileC0[p][col] = (_Float16)(fast_tanh(D2[0][r] + b2) * f0);
        tileC1[p][col] = (_Float16)(fast_tanh(D2[1][r] + b2) * f1);
      }
    } else if (col >= 8 && col < 14) {
#pragma unroll
      for (int r = 0; r < 8; ++r) {
        const int p = r + 8 * hh;
        const float f0 = (float)((uint32_t)(sm[0] >> (8 * r)) & 1u);
        const float f1 = (float)((uint32_t)(sm[1] >> (8 * r)) & 1u);
        tileC0[p][16 + (col - 8)] = (_Float16)(fast_tanh(D2[0][r] + b2) * f0);
        tileC1[p][16 + (col - 8)] = (_Float16)(fast_tanh(D2[1][r] + b2) * f1);
      }
    }

    lo0 = *(const v8h*)&tileC0[col][8 * hh];
    hi0 = *(const v8h*)&tileC0[col][16 + 8 * hh];
    lo1 = *(const v8h*)&tileC1[col][8 * hh];
    hi1 = *(const v8h*)&tileC1[col][16 + 8 * hh];
    v16h A3_0 = __builtin_shufflevector(lo0, hi0, 0,1,2,3,4,5,6,7,8,9,10,11,12,13,14,15);
    v16h A3_1 = __builtin_shufflevector(lo1, hi1, 0,1,2,3,4,5,6,7,8,9,10,11,12,13,14,15);
    v8f D3[UNROLL];
    D3[0] = __builtin_amdgcn_wmma_f32_16x16x32_f16(false, A3_0, false, Bw3, (short)0, zeroC, false, false);
    D3[1] = __builtin_amdgcn_wmma_f32_16x16x32_f16(false, A3_1, false, Bw3, (short)0, zeroC, false, false);

    // ---- tree reduce (rows pre-masked); bias via popcount ----------------
#pragma unroll
    for (int u = 0; u < UNROLL; ++u) {
      const float s01 = D3[u][0] + D3[u][1];
      const float s23 = D3[u][2] + D3[u][3];
      const float s45 = D3[u][4] + D3[u][5];
      const float s67 = D3[u][6] + D3[u][7];
      acc += (s01 + s23) + (s45 + s67);
      if (col < 3)                    cnt += (int)__builtin_popcountll(fm[u]);
      else if (col >= 8 && col < 11)  cnt += (int)__builtin_popcountll(sm[u]);
    }

    afp += APITCH;
    asp += APITCH;
    mp  += MPITCH;
    pp  += MPITCH;
  }

  acc += (float)cnt * b3;  // sum_m mask * b3
  if (col < 3)                    atomicAdd(&pred_sh[col], acc);       // ds_add_f32
  else if (col >= 8 && col < 11)  atomicAdd(&pred_sh[col - 8], acc);
  __syncthreads();
  if (tid < 3) out[(size_t)n * 3 + tid] = pred_sh[tid];
}

// ---------------------------------------------------------------------------
extern "C" void kernel_launch(void* const* d_in, const int* in_sizes, int n_in,
                              void* d_out, int out_size, void* d_ws, size_t ws_size,
                              hipStream_t stream) {
  (void)in_sizes; (void)n_in; (void)out_size; (void)ws_size;
  const uint8_t* mask  = (const uint8_t*)d_in[0];   // bool -> 1 byte/elem
  const float* center  = (const float*)d_in[1];
  const float* cur     = (const float*)d_in[2];
  const float* fW0 = (const float*)d_in[3];  const float* fb0 = (const float*)d_in[4];
  const float* fW1 = (const float*)d_in[5];  const float* fb1 = (const float*)d_in[6];
  const float* fW2 = (const float*)d_in[7];  const float* fb2 = (const float*)d_in[8];
  const float* fW3 = (const float*)d_in[9];  const float* fb3 = (const float*)d_in[10];
  const float* sW0 = (const float*)d_in[11]; const float* sb0 = (const float*)d_in[12];
  const float* sW1 = (const float*)d_in[13]; const float* sb1 = (const float*)d_in[14];
  const float* sW2 = (const float*)d_in[15]; const float* sb2 = (const float*)d_in[16];
  const float* sW3 = (const float*)d_in[17]; const float* sb3 = (const float*)d_in[18];

  float*   Af  = (float*)d_ws;                      // [MM][24] padded
  float*   As  = Af + (size_t)MM * AST;             // [MM][24] padded
  uint8_t* pt8 = (uint8_t*)(As + (size_t)MM * AST); // [MM]
  float*   out = (float*)d_out;

  fluidnet_pre<<<(MM * H0 + 255) / 256, 256, 0, stream>>>(cur, fW0, sW0, Af, As, pt8);
  fluidnet_main<<<NN, 256, 0, stream>>>(mask, center,
                                        fW0, fb0, fW1, fb1, fW2, fb2, fW3, fb3,
                                        sW0, sb0, sW1, sb1, sW2, sb2, sW3, sb3,
                                        Af, As, pt8, out);
}
